// ScaleDotAttention_75514114998374
// MI455X (gfx1250) — matrix operand order repaired and checked by reading, not compiled, not run on hardware
//
#include <hip/hip_runtime.h>
#include <math.h>

// ---------------------------------------------------------------------------
// Flash-attention for B=8, S=2048, D=512 (fp32 in/out), f16 WMMA compute.
// One workgroup (128 threads, 4 wave32) handles one 16-query tile.
// ---------------------------------------------------------------------------

typedef __attribute__((ext_vector_type(16))) _Float16 v16h;
typedef __attribute__((ext_vector_type(8)))  _Float16 v8h;
typedef __attribute__((ext_vector_type(8)))  float    v8f;
typedef __attribute__((ext_vector_type(4)))  float    v4f;

#define B_  8
#define S_  2048
#define D_  512
#define BQ  16
#define BK  32          // keys per outer iteration
#define NKT (S_ / BK)   // 64 iterations
#define QSTR 520        // Qh row stride in halves (16B-multiple, padded)
#define VSTR 40         // Vt row stride in halves (16B-multiple, padded)

__global__ __launch_bounds__(128)
void attn_fa_f16wmma(const float* __restrict__ Q,
                     const float* __restrict__ K,
                     const float* __restrict__ V,
                     const float* __restrict__ Mask,
                     float* __restrict__ Out)
{
    __shared__ __attribute__((aligned(16))) _Float16 Qh[BQ][QSTR];   // 16,640 B
    __shared__ __attribute__((aligned(16))) _Float16 Vt[D_][VSTR];   // 40,960 B (V transposed)
    __shared__ __attribute__((aligned(16))) float    Sc[BQ][BK];     //  2,048 B
    __shared__ __attribute__((aligned(16))) _Float16 Ph[BQ][VSTR];   //  1,280 B
    __shared__ float m_s[BQ];
    __shared__ float l_s[BQ];
    __shared__ float alpha_s[BQ];

    const int tid   = threadIdx.x;
    const int wave  = tid >> 5;
    const int lane  = tid & 31;
    const int lhalf = lane >> 4;     // 0 or 1 (which half-wave)
    const int lmod  = lane & 15;

    const int b  = blockIdx.x >> 7;        // S/16 = 128 query tiles per batch
    const int qt = blockIdx.x & 127;
    const int q0 = qt * BQ;

    const size_t baseQ  = ((size_t)b * S_ + q0) * D_;
    const size_t baseKV = (size_t)b * S_ * D_;

    // ------------------------------------------------------------------
    // Stage Q tile (16 x 512 fp32) into LDS as f16.
    // 2048 float4 loads spread over 128 threads (16 each).
    // ------------------------------------------------------------------
    #pragma unroll
    for (int i = 0; i < 16; ++i) {
        int f   = tid + 128 * i;       // float4 index in 16x128 grid
        int row = f >> 7;              // 128 float4 per row
        int c4  = (f & 127) << 2;
        v4f qv = *(const v4f*)(Q + baseQ + (size_t)row * D_ + c4);
        Qh[row][c4 + 0] = (_Float16)qv[0];
        Qh[row][c4 + 1] = (_Float16)qv[1];
        Qh[row][c4 + 2] = (_Float16)qv[2];
        Qh[row][c4 + 3] = (_Float16)qv[3];
    }
    if (tid < BQ) { m_s[tid] = -INFINITY; l_s[tid] = 0.0f; }

    v8f acc[8] = {};                   // 16 x 128 output accumulator per wave

    __syncthreads();

    const int  kw = wave & 1;          // which 16-key subtile this wave scores
    const int  sw = wave >> 1;         // which half of the D-reduction
    const float scale = 0.04419417382415922f;   // 1/sqrt(512)

    for (int kt = 0; kt < NKT; ++kt) {
        const int k0 = kt * BK;

        // ---- zero score tile (for split-K LDS atomics) -------------------
        #pragma unroll
        for (int i = 0; i < 4; ++i)
            ((float*)Sc)[tid + 128 * i] = 0.0f;

        // ---- stage V tile transposed into LDS: Vt[dim][key] --------------
        // tasks: 16 key-pairs x 128 dim4-groups = 2048, 16 per thread
        #pragma unroll
        for (int i = 0; i < 16; ++i) {
            int task = tid + 128 * i;
            int key  = (task >> 7) << 1;     // even key of the pair
            int d4   = (task & 127) << 2;
            v4f a0 = *(const v4f*)(V + baseKV + (size_t)(k0 + key    ) * D_ + d4);
            v4f a1 = *(const v4f*)(V + baseKV + (size_t)(k0 + key + 1) * D_ + d4);
            #pragma unroll
            for (int j = 0; j < 4; ++j) {
                Vt[d4 + j][key    ] = (_Float16)a0[j];
                Vt[d4 + j][key + 1] = (_Float16)a1[j];
            }
        }
        __syncthreads();

        // =================================================================
        // Phase A: S = (Q K^T) * scale  -- split-K across wave pairs
        // =================================================================
        {
            const int krow = k0 + kw * 16 + lmod;     // this lane's key row (B col)
            const float* Krow = K + baseKV + (size_t)krow * D_;

            v8f sc = {};
            #pragma unroll
            for (int s8 = 0; s8 < 8; ++s8) {
                const int s  = sw * 8 + s8;           // 32-wide K step
                // A fragment (Q) from LDS: row lmod, contiguous k = 16*lhalf + j (same order as B)
                const int d0 = s * 32 + lhalf * 16;
                v8h alo = *(const v8h*)&Qh[lmod][d0];
                v8h ahi = *(const v8h*)&Qh[lmod][d0 + 8];
                v16h a;
                #pragma unroll
                for (int j = 0; j < 8; ++j) { a[j] = alo[j]; a[8 + j] = ahi[j]; }

                // B fragment (K^T) straight from global: 16 contiguous f32
                const int e0 = s * 32 + lhalf * 16;
                v16h bf;
                #pragma unroll
                for (int j = 0; j < 4; ++j) {
                    v4f kv = *(const v4f*)(Krow + e0 + 4 * j);
                    bf[4 * j + 0] = (_Float16)kv[0];
                    bf[4 * j + 1] = (_Float16)kv[1];
                    bf[4 * j + 2] = (_Float16)kv[2];
                    bf[4 * j + 3] = (_Float16)kv[3];
                }
                sc = __builtin_amdgcn_wmma_f32_16x16x32_f16(
                        false, a, false, bf, (short)0, sc, false, false);
            }
            // combine split-K partials in LDS
            #pragma unroll
            for (int j = 0; j < 8; ++j) {
                int row = j + 8 * lhalf;              // C layout: M = vgpr + 8*(lane>=16)
                atomicAdd(&Sc[row][kw * 16 + lmod], sc[j] * scale);
            }
        }
        __syncthreads();

        // =================================================================
        // Online softmax over this 32-key slab (one lane per query row)
        // =================================================================
        if (tid < BQ) {
            const int r = tid;
            float mo = m_s[r];
            float mx = mo;
            #pragma unroll
            for (int c = 0; c < BK; ++c) mx = fmaxf(mx, Sc[r][c]);
            float corr = __expf(mo - mx);             // 0 when mo == -inf
            float lsum = l_s[r] * corr;
            #pragma unroll
            for (int c = 0; c < BK; ++c) {
                float p = __expf(Sc[r][c] - mx);
                Ph[r][c] = (_Float16)p;
                lsum += p;
            }
            m_s[r] = mx; l_s[r] = lsum; alpha_s[r] = corr;
        }
        __syncthreads();

        // =================================================================
        // Phase B: O = alpha*O + P V   (each wave owns 128 output dims)
        // =================================================================
        {
            float al[8];
            #pragma unroll
            for (int j = 0; j < 8; ++j) al[j] = alpha_s[j + 8 * lhalf];

            // A fragment (P, 16x32) from LDS
            v8h plo = *(const v8h*)&Ph[lmod][lhalf * 16];
            v8h phi = *(const v8h*)&Ph[lmod][lhalf * 16 + 8];
            v16h pa;
            #pragma unroll
            for (int j = 0; j < 8; ++j) { pa[j] = plo[j]; pa[8 + j] = phi[j]; }

            #pragma unroll
            for (int nt = 0; nt < 8; ++nt) {
                #pragma unroll
                for (int j = 0; j < 8; ++j) acc[nt][j] *= al[j];

                const int n = wave * 128 + nt * 16 + lmod;   // output dim (B col)
                v8h vlo = *(const v8h*)&Vt[n][lhalf * 16];
                v8h vhi = *(const v8h*)&Vt[n][lhalf * 16 + 8];
                v16h vb;
                #pragma unroll
                for (int j = 0; j < 8; ++j) { vb[j] = vlo[j]; vb[8 + j] = vhi[j]; }

                acc[nt] = __builtin_amdgcn_wmma_f32_16x16x32_f16(
                            false, pa, false, vb, (short)0, acc[nt], false, false);
            }
        }
        __syncthreads();   // protect Vt/Ph/Sc before next iteration
    }

    // ----------------------------------------------------------------------
    // Epilogue: O = (acc / l) * mask[b, q]
    // ----------------------------------------------------------------------
    float invl[8], mk[8];
    #pragma unroll
    for (int j = 0; j < 8; ++j) {
        int row = j + 8 * lhalf;
        invl[j] = 1.0f / l_s[row];
        mk[j]   = Mask[(size_t)b * S_ + q0 + row];
    }
    #pragma unroll
    for (int nt = 0; nt < 8; ++nt) {
        const int n = wave * 128 + nt * 16 + lmod;
        #pragma unroll
        for (int j = 0; j < 8; ++j) {
            int row = j + 8 * lhalf;
            Out[baseQ + (size_t)row * D_ + n] = acc[nt][j] * invl[j] * mk[j];
        }
    }
}

extern "C" void kernel_launch(void* const* d_in, const int* in_sizes, int n_in,
                              void* d_out, int out_size, void* d_ws, size_t ws_size,
                              hipStream_t stream) {
    (void)in_sizes; (void)n_in; (void)out_size; (void)d_ws; (void)ws_size;
    const float* q    = (const float*)d_in[0];
    const float* k    = (const float*)d_in[1];
    const float* v    = (const float*)d_in[2];
    const float* mask = (const float*)d_in[3];
    float* out = (float*)d_out;

    dim3 grid(B_ * (S_ / BQ));   // 1024 workgroups
    dim3 block(128);             // 4 wave32
    hipLaunchKernelGGL(attn_fa_f16wmma, grid, block, 0, stream, q, k, v, mask, out);
}
